// LinearAttention_75617194213719
// MI455X (gfx1250) — matrix operand order repaired and checked by reading, not compiled, not run on hardware
//
#include <hip/hip_runtime.h>

// ---------------------------------------------------------------------------
// Linear attention (non-causal) for MI455X / gfx1250, wave32 + WMMA f16
// + async global->LDS copies (ASYNCcnt) for the bandwidth-critical streams.
//   Phase 1: KV[n,h,m,d] = sum_s V[s,m]*K[s,d],  Ksum[n,h,d] = sum_s K[s,d]
//   Phase 2: Out[n,l,h,m] = (Q[l,:] @ KV[m,:]) / (Q[l,:]·Ksum + eps)
// Memory-bound (~268 MB traffic, ~11.5 us floor @ 23.3 TB/s).
// ---------------------------------------------------------------------------

typedef __attribute__((ext_vector_type(16))) _Float16 v16h;
typedef __attribute__((ext_vector_type(8)))  _Float16 h8;
typedef __attribute__((ext_vector_type(4)))  _Float16 h4;
typedef __attribute__((ext_vector_type(8)))  float    v8f;
typedef __attribute__((ext_vector_type(4)))  int      i4;

typedef __attribute__((address_space(1))) i4 gi4;   // global int4
typedef __attribute__((address_space(3))) i4 li4;   // LDS int4

#define N_ 4
#define L_ 4096
#define H_ 16
#define D_ 64
#define SPLITS 16            // L-splits in phase 1
#define CH 32                // s-chunk per iteration (WMMA K)
#define SSEG (L_ / SPLITS)   // 256 s per phase-1 block
#define P1STRIDE 40          // f16 LDS row stride, 80B -> 16B aligned
#define P2STRIDE 72          // f16 LDS row stride, 144B -> 16B aligned
#define RAWSTRIDE 68         // f32 raw-stage row stride, 272B -> 16B aligned

__device__ __forceinline__ float elu_fm(float x) {
    return x > 0.0f ? x + 1.0f : __expf(x);   // elu(x)+1
}

__device__ __forceinline__ float atomic_add_f32(float* p, float v) {
    return __hip_atomic_fetch_add(p, v, __ATOMIC_RELAXED, __HIP_MEMORY_SCOPE_AGENT);
}

// ---- async global->LDS 16B copy (ASYNCcnt-tracked) -------------------------
__device__ __forceinline__ void async_copy_b128(const float* g, void* l) {
#if __has_builtin(__builtin_amdgcn_global_load_async_to_lds_b128)
    __builtin_amdgcn_global_load_async_to_lds_b128(
        (gi4*)(unsigned long long)g,
        (li4*)(unsigned)(unsigned long long)l, 0, 0);
#else
    // LDS byte offset = low 32 bits of the generic shared address
    unsigned loff = (unsigned)(unsigned long long)l;
    asm volatile("global_load_async_to_lds_b128 %0, %1, off"
                 :: "v"(loff), "v"(g) : "memory");
#endif
}

__device__ __forceinline__ void wait_async_le8() {
#if __has_builtin(__builtin_amdgcn_s_wait_asynccnt)
    __builtin_amdgcn_s_wait_asynccnt(8);
#else
    asm volatile("s_wait_asynccnt 0x8" ::: "memory");
#endif
}

__device__ __forceinline__ void wait_async_0() {
#if __has_builtin(__builtin_amdgcn_s_wait_asynccnt)
    __builtin_amdgcn_s_wait_asynccnt(0);
#else
    asm volatile("s_wait_asynccnt 0x0" ::: "memory");
#endif
}

// ---------------------------------------------------------------------------
__global__ void la_zero(float* __restrict__ p, int n) {
    int i = blockIdx.x * blockDim.x + threadIdx.x;
    if (i < n) p[i] = 0.0f;
}

// ---------------------------------------------------------------------------
// Phase 1: grid = N*H*SPLITS blocks, 128 threads (4 waves).
// Double-buffered async global->LDS of raw f32 K/V chunks, LDS transpose +
// f16 convert, then WMMA. Partials reduced with f32 atomics (land in L2).
// ---------------------------------------------------------------------------
__global__ __launch_bounds__(128) void la_phase1(
    const float* __restrict__ keys, const float* __restrict__ values,
    const float* __restrict__ mask, float* __restrict__ kv,
    float* __restrict__ ksum)
{
    __shared__ __align__(16) float Raw[2][2][CH][RAWSTRIDE]; // [buf][K/V][s][d]
    __shared__ __align__(16) _Float16 Kt[D_][P1STRIDE];      // [d][s_local]
    __shared__ __align__(16) _Float16 Vt[D_][P1STRIDE];      // [m][s_local]
    __shared__ float Msk[SSEG];

    const int bid   = blockIdx.x;
    const int split = bid % SPLITS;
    const int nh    = bid / SPLITS;
    const int h     = nh % H_;
    const int n     = nh / H_;

    const int tid  = threadIdx.x;
    const int lane = tid & 31;
    const int wave = tid >> 5;     // 0..3 -> m block
    const int hl   = lane >> 4;    // lane half
    const int ln   = lane & 15;

    const int s_base = split * SSEG;
    const float* kbase = keys   + (size_t)n * L_ * H_ * D_;
    const float* vbase = values + (size_t)n * L_ * H_ * D_;

    // per-thread float4 slots for a 32x64 chunk (4 slots/thread/matrix)
    int sl[4], c4[4];
    size_t goff[4];
    #pragma unroll
    for (int j = 0; j < 4; ++j) {
        const int t = tid + j * 128;
        sl[j] = t >> 4;                 // 0..31
        c4[j] = (t & 15) * 4;           // 0..60
        goff[j] = ((size_t)(s_base + sl[j]) * H_ + h) * D_ + c4[j];
    }

    // mask slice for this block
    Msk[tid]       = mask[(size_t)n * L_ + s_base + tid];
    Msk[tid + 128] = mask[(size_t)n * L_ + s_base + tid + 128];

    const int iters = SSEG / CH;        // 8

    // prologue: async-issue chunk 0
    #pragma unroll
    for (int j = 0; j < 4; ++j) {
        async_copy_b128(kbase + goff[j], &Raw[0][0][sl[j]][c4[j]]);
        async_copy_b128(vbase + goff[j], &Raw[0][1][sl[j]][c4[j]]);
    }

    const v8f zero8 = {0.f,0.f,0.f,0.f,0.f,0.f,0.f,0.f};
    v8f acc[4] = {zero8, zero8, zero8, zero8};
    float kacc = 0.0f;

    for (int it = 0; it < iters; ++it) {
        // issue next chunk, then wait for the older one (async loads complete
        // in order: <=8 outstanding means chunk `it` is fully in LDS)
        if (it + 1 < iters) {
            const int nb = (it + 1) & 1;
            const size_t add = (size_t)(it + 1) * CH * H_ * D_;
            #pragma unroll
            for (int j = 0; j < 4; ++j) {
                async_copy_b128(kbase + goff[j] + add, &Raw[nb][0][sl[j]][c4[j]]);
                async_copy_b128(vbase + goff[j] + add, &Raw[nb][1][sl[j]][c4[j]]);
            }
            wait_async_le8();
        } else {
            wait_async_0();
        }
        __syncthreads();   // previous iteration's WMMA done reading Kt/Vt

        // ---- LDS-resident transpose + f16 convert (each thread converts the
        //      exact elements its own asyncs delivered)
        const int buf = it & 1;
        #pragma unroll
        for (int j = 0; j < 4; ++j) {
            const float4 kf = *reinterpret_cast<const float4*>(&Raw[buf][0][sl[j]][c4[j]]);
            const float4 vf = *reinterpret_cast<const float4*>(&Raw[buf][1][sl[j]][c4[j]]);
            const float  mk = Msk[it * CH + sl[j]];
            const int c = c4[j], s = sl[j];
            Kt[c + 0][s] = (_Float16)(elu_fm(kf.x) * mk);
            Kt[c + 1][s] = (_Float16)(elu_fm(kf.y) * mk);
            Kt[c + 2][s] = (_Float16)(elu_fm(kf.z) * mk);
            Kt[c + 3][s] = (_Float16)(elu_fm(kf.w) * mk);
            Vt[c + 0][s] = (_Float16)vf.x;
            Vt[c + 1][s] = (_Float16)vf.y;
            Vt[c + 2][s] = (_Float16)vf.z;
            Vt[c + 3][s] = (_Float16)vf.w;
        }
        __syncthreads();

        // ---- Ksum partial (wave-uniform branch), vectorized h8 reads
        if (tid < D_) {
            #pragma unroll
            for (int j = 0; j < 4; ++j) {
                const h8 x = *reinterpret_cast<const h8*>(&Kt[tid][j * 8]);
                #pragma unroll
                for (int e = 0; e < 8; ++e) kacc += (float)x[e];
            }
        }

        // ---- A fragment: A[i=m][k=s] = V^T (ISA 16-bit A 16x32 layout)
        v16h a;
        {
            const int m = wave * 16 + ln;
            const h8 lo = *reinterpret_cast<const h8*>(&Vt[m][hl * 8]);
            const h8 hi = *reinterpret_cast<const h8*>(&Vt[m][16 + hl * 8]);
            #pragma unroll
            for (int e = 0; e < 8; ++e) { a[e] = lo[e]; a[8 + e] = hi[e]; }
        }
        // ---- 4 d-tiles: B[k=s][j=d] = K (ISA B 32x16 layout)
        #pragma unroll
        for (int db = 0; db < 4; ++db) {
            const int d = db * 16 + ln;
            const h8 b0 = *reinterpret_cast<const h8*>(&Kt[d][hl * 8]);
            const h8 b1 = *reinterpret_cast<const h8*>(&Kt[d][16 + hl * 8]);
            v16h b;
            #pragma unroll
            for (int e = 0; e < 8; ++e) { b[e] = b0[e]; b[8 + e] = b1[e]; }
            acc[db] = __builtin_amdgcn_wmma_f32_16x16x32_f16(
                false, a, false, b, (short)0, acc[db], false, false);
        }
    }

    // ---- reduce partials into global KV / Ksum via f32 atomics
    float* kvp = kv + (size_t)(n * H_ + h) * D_ * D_;
    #pragma unroll
    for (int db = 0; db < 4; ++db) {
        const int d = db * 16 + ln;
        #pragma unroll
        for (int r = 0; r < 8; ++r) {
            const int m = wave * 16 + hl * 8 + r;   // C/D layout: M = r (+8 hi half)
            atomic_add_f32(&kvp[m * D_ + d], acc[db][r]);
        }
    }
    if (tid < D_)
        atomic_add_f32(&ksum[(size_t)(n * H_ + h) * D_ + tid], kacc);
}

// ---------------------------------------------------------------------------
// Phase 2: grid = N*H*(L/128) blocks, 256 threads (8 waves).
// Q tile is async-prefetched into LDS (raw f32) while KV (L2-resident) is
// staged; each wave computes a 16(l) x 64(m) tile: Q @ KV^T, scaled by Z.
// ---------------------------------------------------------------------------
__global__ __launch_bounds__(256) void la_phase2(
    const float* __restrict__ queries, const float* __restrict__ kv,
    const float* __restrict__ ksum, float* __restrict__ out)
{
    __shared__ __align__(16) float    Qraw[128][RAWSTRIDE];  // async landing pad
    __shared__ __align__(16) _Float16 KVs[D_][P2STRIDE];     // [m][d]
    __shared__ __align__(16) _Float16 Qs[128][P2STRIDE];     // [l_local][d]
    __shared__ float Ks[D_];
    __shared__ float Zs[128];

    const int LBLK = L_ / 128;          // 32 blocks per (n,h)
    const int bid  = blockIdx.x;
    const int lb   = bid % LBLK;
    const int nh   = bid / LBLK;
    const int h    = nh % H_;
    const int n    = nh / H_;
    const int l0   = lb * 128;

    const int tid  = threadIdx.x;
    const int lane = tid & 31;
    const int wave = tid >> 5;          // 0..7 -> l block of 16
    const int hl   = lane >> 4;
    const int ln   = lane & 15;

    // ---- async-prefetch the 128x64 f32 Q tile (8 x b128 per thread)
    int ql[8], qc[8];
    #pragma unroll
    for (int j = 0; j < 8; ++j) {
        const int t = tid + j * 256;
        ql[j] = t >> 4;
        qc[j] = (t & 15) * 4;
        const float* g = queries + (((size_t)n * L_ + l0 + ql[j]) * H_ + h) * D_ + qc[j];
        async_copy_b128(g, &Qraw[ql[j]][qc[j]]);
    }

    // ---- meanwhile stage KV (f32 -> f16, lives in L2) and Ksum
    const float* kvp = kv + (size_t)(n * H_ + h) * D_ * D_;
    #pragma unroll
    for (int j = 0; j < 4; ++j) {
        const int t  = tid + j * 256;
        const int m  = t >> 4;
        const int c4 = (t & 15) * 4;
        const float4 f = *reinterpret_cast<const float4*>(kvp + m * D_ + c4);
        h4 hv = { (_Float16)f.x, (_Float16)f.y, (_Float16)f.z, (_Float16)f.w };
        *reinterpret_cast<h4*>(&KVs[m][c4]) = hv;
    }
    if (tid < D_) Ks[tid] = ksum[(size_t)(n * H_ + h) * D_ + tid];

    // ---- wait for own asyncs, convert Q (elu+1, f32 -> f16)
    wait_async_0();
    #pragma unroll
    for (int j = 0; j < 8; ++j) {
        const float4 f = *reinterpret_cast<const float4*>(&Qraw[ql[j]][qc[j]]);
        h4 hv = { (_Float16)elu_fm(f.x), (_Float16)elu_fm(f.y),
                  (_Float16)elu_fm(f.z), (_Float16)elu_fm(f.w) };
        *reinterpret_cast<h4*>(&Qs[ql[j]][qc[j]]) = hv;
    }
    __syncthreads();

    // ---- Z[l] = 1/(Q[l,:]·Ksum + eps)  (wave-uniform branch)
    if (tid < 128) {
        float zz = 0.0f;
        #pragma unroll
        for (int d = 0; d < D_; ++d) zz += (float)Qs[tid][d] * Ks[d];
        Zs[tid] = 1.0f / (zz + 1e-6f);
    }
    __syncthreads();

    // ---- GEMM: D[l][m] = sum_d Q[l][d] * KV[m][d]
    const v8f zero8 = {0.f,0.f,0.f,0.f,0.f,0.f,0.f,0.f};
    v8f acc[4] = {zero8, zero8, zero8, zero8};
    #pragma unroll
    for (int dc = 0; dc < 2; ++dc) {        // d chunks of 32
        v16h a;
        {
            const int l = wave * 16 + ln;
            const h8 lo = *reinterpret_cast<const h8*>(&Qs[l][dc * 32 + hl * 8]);
            const h8 hi = *reinterpret_cast<const h8*>(&Qs[l][dc * 32 + 16 + hl * 8]);
            #pragma unroll
            for (int e = 0; e < 8; ++e) { a[e] = lo[e]; a[8 + e] = hi[e]; }
        }
        #pragma unroll
        for (int mb = 0; mb < 4; ++mb) {
            const int m = mb * 16 + ln;     // B[k=d][j=m] = KV[m][d]
            const h8 b0 = *reinterpret_cast<const h8*>(&KVs[m][dc * 32 + hl * 8]);
            const h8 b1 = *reinterpret_cast<const h8*>(&KVs[m][dc * 32 + 16 + hl * 8]);
            v16h b;
            #pragma unroll
            for (int e = 0; e < 8; ++e) { b[e] = b0[e]; b[8 + e] = b1[e]; }
            acc[mb] = __builtin_amdgcn_wmma_f32_16x16x32_f16(
                false, a, false, b, (short)0, acc[mb], false, false);
        }
    }

    // ---- scale by Z and store (coalesced f32 per 16-lane group)
    #pragma unroll
    for (int mb = 0; mb < 4; ++mb) {
        const int m = mb * 16 + ln;
        #pragma unroll
        for (int r = 0; r < 8; ++r) {
            const int ll = wave * 16 + hl * 8 + r;
            out[(((size_t)n * L_ + l0 + ll) * H_ + h) * D_ + m] = acc[mb][r] * Zs[ll];
        }
    }
}

// ---------------------------------------------------------------------------
extern "C" void kernel_launch(void* const* d_in, const int* in_sizes, int n_in,
                              void* d_out, int out_size, void* d_ws, size_t ws_size,
                              hipStream_t stream) {
    const float* q    = (const float*)d_in[0];
    const float* k    = (const float*)d_in[1];
    const float* v    = (const float*)d_in[2];
    const float* mask = (const float*)d_in[3];
    float* out = (float*)d_out;

    float* kv = (float*)d_ws;                                  // N*H*64*64 f32
    float* ks = kv + (size_t)N_ * H_ * D_ * D_;                // N*H*64 f32

    const int ztot = N_ * H_ * D_ * D_ + N_ * H_ * D_;
    la_zero<<<(ztot + 255) / 256, 256, 0, stream>>>(kv, ztot);
    la_phase1<<<N_ * H_ * SPLITS, 128, 0, stream>>>(k, v, mask, kv, ks);
    la_phase2<<<N_ * H_ * (L_ / 128), 256, 0, stream>>>(q, kv, ks, out);
}
